// SRNN_60481729462520
// MI455X (gfx1250) — matrix operand order repaired and checked by reading, not compiled, not run on hardware
//
#include <hip/hip_runtime.h>
#include <hip/hip_bf16.h>
#include <math.h>

// ---------------- problem constants ----------------
#define NT   500
#define NB   128
#define NIN  700
#define NINP 704          // 700 padded to multiple of 32
#define NREC 2048
#define NOUT 20
#define NOUTP 32          // W_out rows padded to 32 (zero rows 20..31)
#define ALPHA 0.95122942450071400909f   // exp(-1e-3/20e-3)
#define KAPPA 0.95122942450071400909f
#define THR   0.6f

typedef __bf16 bf16_t;
typedef __bf16 bf16x8   __attribute__((ext_vector_type(8)));
typedef __bf16 bf16x16  __attribute__((ext_vector_type(16)));
typedef float  f32x8    __attribute__((ext_vector_type(8)));

// ---- gfx1250 async global->LDS path (builtin confirmed present by probe) ----
#if __has_builtin(__builtin_amdgcn_global_load_async_to_lds_b128)
#define HAS_ASYNC_LDS 1
// builtin signature: (int __vector(4) AS1*, int __vector(4) AS3*, imm offset, imm cpol)
typedef int v4i_vs __attribute__((__vector_size__(16)));
using gv4i_p = __attribute__((address_space(1))) v4i_vs*;
using lv4i_p = __attribute__((address_space(3))) v4i_vs*;
#endif

// stage 16 bytes into LDS (async if available, plain ds store otherwise)
static __device__ inline void stage16(bf16_t* dst, const bf16_t* src) {
#if defined(HAS_ASYNC_LDS)
  __builtin_amdgcn_global_load_async_to_lds_b128(
      (gv4i_p)(uintptr_t)src, (lv4i_p)(uintptr_t)dst, 0, 0);
#else
  *(bf16x8*)dst = *(const bf16x8*)src;
#endif
}

// make staged data visible block-wide
static __device__ inline void stage_commit() {
#if defined(HAS_ASYNC_LDS)
#if __has_builtin(__builtin_amdgcn_s_wait_asynccnt)
  __builtin_amdgcn_s_wait_asynccnt(0);
#else
  asm volatile("s_wait_asynccnt 0x0" ::: "memory");
#endif
#endif
  __syncthreads();
}

// ---------------- WMMA fragment helpers (wave32, 16x16x32 bf16) ----------------

static __device__ inline bf16x16 concat8(bf16x8 g0, bf16x8 g1) {
  return __builtin_shufflevector(g0, g1, 0,1,2,3,4,5,6,7,8,9,10,11,12,13,14,15);
}

// A matrix 16x32 bf16: lane m=lane&15; lanes 0-15 hold K = {0..7,16..23},
// lanes 16-31 hold K = {8..15,24..31}  (ISA 7.12.2, 16-bit A layout)
static __device__ inline bf16x16 load_a_bf16(const bf16_t* __restrict__ base,
                                             int row_stride, int row0, int k0, int lane) {
  const int m  = lane & 15;
  const int kg = k0 + ((lane < 16) ? 0 : 8);
  const bf16_t* p = base + (size_t)(row0 + m) * row_stride + kg;
  bf16x8 g0 = *(const bf16x8*)(p);
  bf16x8 g1 = *(const bf16x8*)(p + 16);
  return concat8(g0, g1);
}

// A fragment sourced from fp32 x with K guard (converts to bf16 in-register)
static __device__ inline bf16x16 load_a_x(const float* __restrict__ xr,
                                          int b0, int k0, int lane) {
  const int m  = lane & 15;
  const int ka = k0 + ((lane < 16) ? 0 : 8);
  const int kb = ka + 16;
  const float* r = xr + (size_t)(b0 + m) * NIN;
  bf16x16 a;
#pragma unroll
  for (int i = 0; i < 8; ++i) {
    float f0 = (ka + i < NIN) ? r[ka + i] : 0.0f;
    float f1 = (kb + i < NIN) ? r[kb + i] : 0.0f;
    a[i]     = (bf16_t)f0;
    a[8 + i] = (bf16_t)f1;
  }
  return a;
}

// B matrix 32x16 bf16: lane holds column n=lane&15; lanes 0-15 K={0..7,16..23},
// lanes 16-31 K={8..15,24..31} (matches the A layout).
static __device__ inline bf16x16 load_b_lds(const bf16_t* __restrict__ lds,
                                            int rrow0, int lane) {
  const int n = lane & 15;
  const bf16_t* p = lds + (rrow0 + n) * 32 + ((lane < 16) ? 0 : 8);
  bf16x8 g0 = *(const bf16x8*)(p);
  bf16x8 g1 = *(const bf16x8*)(p + 16);
  return concat8(g0, g1);
}

// unguarded global B load (W_out is zero-padded to 32 rows)
static __device__ inline bf16x16 load_b_global(const bf16_t* __restrict__ base,
                                               int row_stride, int col0, int k0, int lane) {
  const int n = lane & 15;
  const bf16_t* p = base + (size_t)(col0 + n) * row_stride + k0 + ((lane < 16) ? 0 : 8);
  bf16x8 g0 = *(const bf16x8*)(p);
  bf16x8 g1 = *(const bf16x8*)(p + 16);
  return concat8(g0, g1);
}

static __device__ inline f32x8 wmma_bb(bf16x16 a, bf16x16 b, f32x8 c) {
  return __builtin_amdgcn_wmma_f32_16x16x32_bf16(
      /*neg_a=*/false, a, /*neg_b=*/false, b,
      /*c_mod=*/(short)0, c, /*reuse_a=*/false, /*reuse_b=*/false);
}

// ---------------- prep kernels (run once, weights stay L2-resident) ----------------

__global__ void srnn_prep_wrec(const float* __restrict__ w, bf16_t* __restrict__ o) {
  for (size_t i = (size_t)blockIdx.x * blockDim.x + threadIdx.x;
       i < (size_t)NREC * NREC; i += (size_t)gridDim.x * blockDim.x) {
    const int r = (int)(i >> 11), k = (int)(i & (NREC - 1));
    o[i] = (bf16_t)((r == k) ? 0.0f : w[i]);   // zero recurrent self-connections
  }
}

__global__ void srnn_prep_win(const float* __restrict__ w, bf16_t* __restrict__ o) {
  for (size_t i = (size_t)blockIdx.x * blockDim.x + threadIdx.x;
       i < (size_t)NREC * NINP; i += (size_t)gridDim.x * blockDim.x) {
    const int r = (int)(i / NINP), k = (int)(i - (size_t)r * NINP);
    o[i] = (k < NIN) ? (bf16_t)w[(size_t)r * NIN + k] : (bf16_t)0.0f;
  }
}

__global__ void srnn_prep_wout(const float* __restrict__ w, bf16_t* __restrict__ o) {
  for (size_t i = (size_t)blockIdx.x * blockDim.x + threadIdx.x;
       i < (size_t)NOUTP * NREC; i += (size_t)gridDim.x * blockDim.x) {
    const int r = (int)(i >> 11), k = (int)(i & (NREC - 1));
    o[i] = (r < NOUT) ? (bf16_t)w[(size_t)r * NREC + k] : (bf16_t)0.0f;
  }
}

__global__ void srnn_prep_state(bf16_t* __restrict__ z0, bf16_t* __restrict__ z1,
                                float* __restrict__ v, float* __restrict__ vo) {
  for (size_t i = (size_t)blockIdx.x * blockDim.x + threadIdx.x;
       i < (size_t)NB * NREC; i += (size_t)gridDim.x * blockDim.x) {
    z0[i] = (bf16_t)0.0f;
    z1[i] = (bf16_t)0.0f;
    v[i]  = 0.0f;
    if (i < (size_t)NB * NOUT) vo[i] = 0.0f;
  }
}

// ---------------- fused per-timestep kernel ----------------
// Blocks 0..31 : v/z update (each block owns a 128x64 strip; 8 waves x 4 col tiles)
// Blocks 32..33: readout vo_t = KAPPA*vo + z_t @ W_out.T  (uses *previous* launch's z)

__global__ __launch_bounds__(256) void srnn_step(
    const float*  __restrict__ x,      // [NT][NB][NIN]
    const bf16_t* __restrict__ wrecB,  // [NREC][NREC] bf16, diag=0
    const bf16_t* __restrict__ winB,   // [NREC][NINP] bf16
    const bf16_t* __restrict__ woutB,  // [NOUTP][NREC] bf16, rows 20..31 zero
    const bf16_t* __restrict__ zR,     // spikes z_t  (read)
    bf16_t*       __restrict__ zW,     // spikes z_{t+1} (write)
    float*        __restrict__ v,      // [NB][NREC]
    float*        __restrict__ vo,     // [NB][NOUT]
    float*        __restrict__ out,    // [NT][NB][NOUT] raw vo (softmax later)
    int t)
{
  __shared__ bf16_t ldsB[2][64 * 32];  // double-buffered 64-col x 32-K B tile
  const int lane = threadIdx.x & 31;
  const int wid  = threadIdx.x >> 5;

  if (blockIdx.x < 32) {
    if (t >= NT - 1) return;           // last launch only runs the readout part
    const int rcol0 = blockIdx.x * 64; // this block's 64 recurrent columns
    const int b0    = wid * 16;        // this wave's 16 batch rows
    const int srow  = threadIdx.x >> 2;        // LDS stage: 64 rows x 4 chunks
    const int sch   = (threadIdx.x & 3) * 8;
    const bf16_t* wrecRow = wrecB + (size_t)(rcol0 + srow) * NREC + sch;
    const bf16_t* winRow  = winB  + (size_t)(rcol0 + srow) * NINP + sch;
    bf16_t* slot0 = &ldsB[0][srow * 32 + sch];
    bf16_t* slot1 = &ldsB[1][srow * 32 + sch];

    f32x8 acc0 = {0,0,0,0,0,0,0,0}, acc1 = {0,0,0,0,0,0,0,0};
    f32x8 acc2 = {0,0,0,0,0,0,0,0}, acc3 = {0,0,0,0,0,0,0,0};

    // ---- z_t @ W_rec.T : K = 2048, 64 iterations, software-pipelined ----
    stage16(slot0, wrecRow);
    stage_commit();
    const int NKR = NREC / 32;
    for (int i = 0; i < NKR; ++i) {
      const int k0 = i * 32;
      if (i + 1 < NKR) {
        stage16((i & 1) ? slot0 : slot1, wrecRow + (k0 + 32));
        if (k0 + 64 < NREC) __builtin_prefetch(wrecRow + (k0 + 64), 0, 1);
      }
      const bf16_t* buf = ldsB[i & 1];
      bf16x16 a = load_a_bf16(zR, NREC, b0, k0, lane);
      acc0 = wmma_bb(a, load_b_lds(buf,  0, lane), acc0);
      acc1 = wmma_bb(a, load_b_lds(buf, 16, lane), acc1);
      acc2 = wmma_bb(a, load_b_lds(buf, 32, lane), acc2);
      acc3 = wmma_bb(a, load_b_lds(buf, 48, lane), acc3);
      if (i + 1 < NKR) stage_commit();
    }

    // ---- x_t @ W_in.T : K = 704 (padded), 22 iterations ----
    // NKR is even, so the last rec tile lived in ldsB[1]; buffer 0 is free.
    const float* xt = x + (size_t)t * NB * NIN;
    stage16(slot0, winRow);
    stage_commit();
    const int NKI = NINP / 32;
    for (int i = 0; i < NKI; ++i) {
      const int k0 = i * 32;
      if (i + 1 < NKI) stage16((i & 1) ? slot0 : slot1, winRow + (k0 + 32));
      const bf16_t* buf = ldsB[i & 1];
      bf16x16 a = load_a_x(xt, b0, k0, lane);
      acc0 = wmma_bb(a, load_b_lds(buf,  0, lane), acc0);
      acc1 = wmma_bb(a, load_b_lds(buf, 16, lane), acc1);
      acc2 = wmma_bb(a, load_b_lds(buf, 32, lane), acc2);
      acc3 = wmma_bb(a, load_b_lds(buf, 48, lane), acc3);
      if (i + 1 < NKI) stage_commit();
    }

    // ---- LIF update + spike ----
    const int n     = lane & 15;
    const int mbase = (lane < 16) ? 0 : 8;
#pragma unroll
    for (int ct = 0; ct < 4; ++ct) {
      f32x8 acc = (ct == 0) ? acc0 : (ct == 1) ? acc1 : (ct == 2) ? acc2 : acc3;
      const int r = rcol0 + ct * 16 + n;
#pragma unroll
      for (int j = 0; j < 8; ++j) {
        const int    b   = b0 + mbase + j;
        const size_t idx = (size_t)b * NREC + r;
        const float  zo  = (float)zR[idx];
        const float  vn  = ALPHA * v[idx] + acc[j] - THR * zo;
        v[idx]  = vn;
        zW[idx] = (bf16_t)((vn > THR) ? 1.0f : 0.0f);
      }
    }
  } else {
    // ---- readout: vo = KAPPA*vo + z_t @ W_out.T, write out[t] ----
    const int o0 = (blockIdx.x - 32) * 16;   // 0 or 16 (cols 20..31 are zero pad)
    const int b0 = wid * 16;
    f32x8 acc = {0,0,0,0,0,0,0,0};
    for (int k0 = 0; k0 < NREC; k0 += 32) {
      bf16x16 a  = load_a_bf16(zR, NREC, b0, k0, lane);
      bf16x16 bf = load_b_global(woutB, NREC, o0, k0, lane);
      acc = wmma_bb(a, bf, acc);
    }
    const int n = lane & 15;
    const int o = o0 + n;
    if (o < NOUT) {
      const int mbase = (lane < 16) ? 0 : 8;
#pragma unroll
      for (int j = 0; j < 8; ++j) {
        const int    b   = b0 + mbase + j;
        const size_t idx = (size_t)b * NOUT + o;
        const float  vn  = KAPPA * vo[idx] + acc[j];
        vo[idx] = vn;
        out[(size_t)t * NB * NOUT + idx] = vn;
      }
    }
  }
}

// ---------------- softmax over the 20 outputs, in-place ----------------
__global__ void srnn_softmax(float* __restrict__ out) {
  const int row = blockIdx.x * blockDim.x + threadIdx.x;
  if (row >= NT * NB) return;
  float* p = out + (size_t)row * NOUT;
  float mx = p[0];
#pragma unroll
  for (int i = 1; i < NOUT; ++i) mx = fmaxf(mx, p[i]);
  float e[NOUT], s = 0.0f;
#pragma unroll
  for (int i = 0; i < NOUT; ++i) { e[i] = __expf(p[i] - mx); s += e[i]; }
  const float inv = 1.0f / s;
#pragma unroll
  for (int i = 0; i < NOUT; ++i) p[i] = e[i] * inv;
}

// ---------------- launch ----------------
extern "C" void kernel_launch(void* const* d_in, const int* in_sizes, int n_in,
                              void* d_out, int out_size, void* d_ws, size_t ws_size,
                              hipStream_t stream) {
  (void)in_sizes; (void)n_in; (void)out_size; (void)ws_size;
  const float* x     = (const float*)d_in[0];
  // d_in[1] = yt (unused in forward), d_in[5] = do_training (unused)
  const float* w_in  = (const float*)d_in[2];
  const float* w_rec = (const float*)d_in[3];
  const float* w_out = (const float*)d_in[4];
  float* out = (float*)d_out;

  char* ws = (char*)d_ws;
  size_t off = 0;
  auto carve = [&](size_t bytes) -> void* {
    void* p = ws + off;
    off += (bytes + 255) & ~(size_t)255;
    return p;
  };
  bf16_t* wrecB = (bf16_t*)carve((size_t)NREC * NREC * sizeof(bf16_t));  // 8 MB
  bf16_t* winB  = (bf16_t*)carve((size_t)NREC * NINP * sizeof(bf16_t));  // 2.9 MB
  bf16_t* woutB = (bf16_t*)carve((size_t)NOUTP * NREC * sizeof(bf16_t)); // 128 KB
  bf16_t* z0    = (bf16_t*)carve((size_t)NB * NREC * sizeof(bf16_t));
  bf16_t* z1    = (bf16_t*)carve((size_t)NB * NREC * sizeof(bf16_t));
  float*  v     = (float*)carve((size_t)NB * NREC * sizeof(float));
  float*  vo    = (float*)carve((size_t)NB * NOUT * sizeof(float));

  srnn_prep_wrec <<<1024, 256, 0, stream>>>(w_rec, wrecB);
  srnn_prep_win  <<<512,  256, 0, stream>>>(w_in,  winB);
  srnn_prep_wout <<<64,   256, 0, stream>>>(w_out, woutB);
  srnn_prep_state<<<512,  256, 0, stream>>>(z0, z1, v, vo);

  for (int t = 0; t < NT; ++t) {
    bf16_t* zR = (t & 1) ? z1 : z0;
    bf16_t* zW = (t & 1) ? z0 : z1;
    srnn_step<<<34, 256, 0, stream>>>(x, wrecB, winB, woutB, zR, zW, v, vo, out, t);
  }

  srnn_softmax<<<(NT * NB + 255) / 256, 256, 0, stream>>>(out);
}
